// DeepRNN_52080773431928
// MI455X (gfx1250) — compile-verified, matrix operand order checked by reading, not run on hardware
//
#include <hip/hip_runtime.h>

typedef __attribute__((ext_vector_type(16))) __bf16 v16bf;
typedef __attribute__((ext_vector_type(8)))  __bf16 v8bf;
typedef __attribute__((ext_vector_type(8)))  float  v8f;

#define GLOBAL_AS __attribute__((address_space(1)))
typedef GLOBAL_AS const __bf16* gbf_p;   // global-addrspace bf16 pointer
typedef GLOBAL_AS const v8bf*  gv8_p;
typedef GLOBAL_AS const float* gf_p;
typedef GLOBAL_AS float*       gfw_p;

#define RNN_B 64
#define RNN_T 1024
#define RNN_I 256
#define RNN_H 512
#define TILE_M 16
#define BLOCK  512
#define NWAVES 16
#define COLS_PER_WAVE 32   /* RNN_H / NWAVES */

// ---------- fast tanh: gfx1250 has V_TANH_F32 in the TRANS unit ----------
__device__ __forceinline__ float tanh_fast(float x) {
#if __has_builtin(__builtin_amdgcn_tanhf)
  return __builtin_amdgcn_tanhf(x);
#elif __has_builtin(__builtin_amdgcn_tanh_f32)
  return __builtin_amdgcn_tanh_f32(x);
#else
  float e = __expf(2.0f * x);           // v_exp_f32 path
  return 1.0f - 2.0f / (e + 1.0f);
#endif
}

// Scheduling fence: keep prefetch loads above the WMMAs of the current
// pipeline stage (the scheduler otherwise sinks loads to their uses and
// destroys the pipeline -> s_wait_loadcnt 0x0 before every WMMA).
__device__ __forceinline__ void sched_split() {
#if __has_builtin(__builtin_amdgcn_sched_barrier)
  __builtin_amdgcn_sched_barrier(0);
#endif
}

// CDNA5 16-bit operand layout: lane = m (+ half*16); VGPR0..3 hold
// K = half*8+0..7, VGPR4..7 hold K = 16+half*8+0..7 -> two contiguous
// 16B chunks at +0 and +16 elements. Caller passes rowbase + half*8.
__device__ __forceinline__ v16bf load_frag_lds(const __bf16* p) {
  v8bf lo = *(const v8bf*)(p);
  v8bf hi = *(const v8bf*)(p + 16);
  return __builtin_shufflevector(lo, hi, 0, 1, 2, 3, 4, 5, 6, 7,
                                 8, 9, 10, 11, 12, 13, 14, 15);
}

__device__ __forceinline__ v16bf load_frag_global(gbf_p p) {
  v8bf lo = *(gv8_p)(p);
  v8bf hi = *(gv8_p)(p + 16);
  return __builtin_shufflevector(lo, hi, 0, 1, 2, 3, 4, 5, 6, 7,
                                 8, 9, 10, 11, 12, 13, 14, 15);
}

__device__ __forceinline__ v8f wmma_bf16(v16bf a, v16bf b, v8f c) {
  return __builtin_amdgcn_wmma_f32_16x16x32_bf16(false, a, false, b,
                                                 (short)0, c, false, false);
}

// Fully-unrolled, 3-stage software-pipelined GEMM accumulate over NK
// k-tiles of 32: loads for kt+3 are issued (and fenced) before the WMMAs
// of kt. Full unroll makes every stage a distinct SSA value (no rotation
// copies) and folds all addresses into immediate offsets.
template <int NK>
__device__ __forceinline__ void gemm_acc(const __bf16* arow,
                                         gbf_p b0base, gbf_p b1base,
                                         v8f& acc0, v8f& acc1) {
  constexpr int PIPE = (NK < 3) ? NK : 3;
  v16bf a[NK], b0[NK], b1[NK];
#pragma unroll
  for (int s = 0; s < PIPE; ++s) {
    a[s]  = load_frag_lds(arow + s * 32);
    b0[s] = load_frag_global(b0base + s * 32);
    b1[s] = load_frag_global(b1base + s * 32);
  }
#pragma unroll
  for (int kt = 0; kt < NK; ++kt) {
    if (kt + PIPE < NK) {
      a[kt + PIPE]  = load_frag_lds(arow + (kt + PIPE) * 32);
      b0[kt + PIPE] = load_frag_global(b0base + (kt + PIPE) * 32);
      b1[kt + PIPE] = load_frag_global(b1base + (kt + PIPE) * 32);
      sched_split();
    }
    acc0 = wmma_bf16(a[kt], b0[kt], acc0);
    acc1 = wmma_bf16(a[kt], b1[kt], acc1);
  }
}

// ---------- weight prep: fp32 [K][N] -> bf16 [N][K] (per-column contiguous) ----------
__global__ __launch_bounds__(256) void transpose_to_bf16(
    const float* __restrict__ W, unsigned short* __restrict__ WT, int K, int N) {
  int idx = blockIdx.x * 256 + threadIdx.x;
  if (idx >= K * N) return;
  int n = idx / K;
  int k = idx - n * K;
  __bf16 v = (__bf16)W[(long)k * N + n];
  WT[idx] = __builtin_bit_cast(unsigned short, v);
}

// ---------- persistent RNN: one workgroup per 16-row batch tile ----------
__global__ __launch_bounds__(BLOCK) void rnn_persistent(
    const float* __restrict__ x,             // [B,T,I] fp32
    const float* __restrict__ bh0,           // [H]
    const float* __restrict__ bh1,           // [H]
    const unsigned short* __restrict__ wxt0, // bf16 [H][I]
    const unsigned short* __restrict__ wht0, // bf16 [H][H]
    const unsigned short* __restrict__ wxt1, // bf16 [H][H]
    const unsigned short* __restrict__ wht1, // bf16 [H][H]
    float* __restrict__ out)                 // [B,T,H] fp32
{
  __shared__ __align__(16) __bf16 lds_x[TILE_M * RNN_I];   // 8 KB
  __shared__ __align__(16) __bf16 lds_h0[TILE_M * RNN_H];  // 16 KB
  __shared__ __align__(16) __bf16 lds_h1[TILE_M * RNN_H];  // 16 KB

  const int tid   = threadIdx.x;
  const int wave  = tid >> 5;
  const int lane  = tid & 31;
  const int m     = lane & 15;
  const int half  = lane >> 4;
  const int brow0 = blockIdx.x * TILE_M;

  // h0 = h1 = 0
  for (int i = tid; i < TILE_M * RNN_H; i += BLOCK) {
    lds_h0[i] = (__bf16)0.0f;
    lds_h1[i] = (__bf16)0.0f;
  }

  const int col0 = wave * COLS_PER_WAVE + m;  // first column this lane owns
  const int col1 = col0 + 16;                 // second tile's column
  float bias0a = bh0[col0], bias0b = bh0[col1];
  float bias1a = bh1[col0], bias1b = bh1[col1];
  __syncthreads();

  // x staging: 32 threads per batch row, 8 floats per thread
  const int xrow = tid >> 5;
  const int xcol = (tid & 31) * 8;
  const float* xsrc0 = x + (long)(brow0 + xrow) * RNN_T * RNN_I + xcol;
  gf_p  gxsrc0 = __builtin_bit_cast(gf_p, xsrc0);
  gfw_p gout   = __builtin_bit_cast(gfw_p, out);

  // Global-addrspace weight pointers, laundered each timestep so the
  // compiler cannot hoist (and then spill) the weight loads out of the
  // 1024-iteration recurrence: each step must stream from L2 via
  // global_load_b128 (LOADcnt only, decoupled from the LDS DScnt path).
  gbf_p WxT0 = __builtin_bit_cast(gbf_p, (const __bf16*)wxt0);
  gbf_p WhT0 = __builtin_bit_cast(gbf_p, (const __bf16*)wht0);
  gbf_p WxT1 = __builtin_bit_cast(gbf_p, (const __bf16*)wxt1);
  gbf_p WhT1 = __builtin_bit_cast(gbf_p, (const __bf16*)wht1);

  for (int t = 0; t < RNN_T; ++t) {
    // opaque per-iteration redefinition of the weight bases (zero cost)
    asm volatile("" : "+s"(WxT0), "+s"(WhT0), "+s"(WxT1), "+s"(WhT1));

    // ---- stage x_t tile fp32 -> bf16 into LDS; prefetch x_{t+1} ----
    {
      const float* psrc = xsrc0 + (long)t * RNN_I;
      if (t + 1 < RNN_T) __builtin_prefetch(psrc + RNN_I, 0, 0);
      gf_p src = gxsrc0 + (long)t * RNN_I;
      v8bf p0;
#pragma unroll
      for (int i = 0; i < 8; ++i) p0[i] = (__bf16)src[i];
      *(v8bf*)&lds_x[xrow * RNN_I + xcol] = p0;
    }
    __syncthreads();

    // ---- layer 0: acc = x_t @ Wxh0 + h0 @ Whh0 + b0 ----
    v8f acc0, acc1;
#pragma unroll
    for (int r = 0; r < 8; ++r) { acc0[r] = bias0a; acc1[r] = bias0b; }

    gemm_acc<RNN_I / 32>(lds_x + m * RNN_I + half * 8,
                         WxT0 + (long)col0 * RNN_I + half * 8,
                         WxT0 + (long)col1 * RNN_I + half * 8, acc0, acc1);
    gemm_acc<RNN_H / 32>(lds_h0 + m * RNN_H + half * 8,
                         WhT0 + (long)col0 * RNN_H + half * 8,
                         WhT0 + (long)col1 * RNN_H + half * 8, acc0, acc1);
    __syncthreads();  // every wave finished reading lds_h0 / lds_x

    // h0_new = tanh(acc); D layout: lane owns column colX, rows half*8+r
#pragma unroll
    for (int r = 0; r < 8; ++r) {
      int row = half * 8 + r;
      lds_h0[row * RNN_H + col0] = (__bf16)tanh_fast(acc0[r]);
      lds_h0[row * RNN_H + col1] = (__bf16)tanh_fast(acc1[r]);
    }
    __syncthreads();  // h0_new visible to all waves

    // ---- layer 1: acc = h0_new @ Wxh1 + h1 @ Whh1 + b1 ----
#pragma unroll
    for (int r = 0; r < 8; ++r) { acc0[r] = bias1a; acc1[r] = bias1b; }

    gemm_acc<RNN_H / 32>(lds_h0 + m * RNN_H + half * 8,
                         WxT1 + (long)col0 * RNN_H + half * 8,
                         WxT1 + (long)col1 * RNN_H + half * 8, acc0, acc1);
    gemm_acc<RNN_H / 32>(lds_h1 + m * RNN_H + half * 8,
                         WhT1 + (long)col0 * RNN_H + half * 8,
                         WhT1 + (long)col1 * RNN_H + half * 8, acc0, acc1);
    __syncthreads();  // every wave finished reading lds_h1

    // h1_new = tanh(acc) -> LDS (bf16, for next step) + global out (fp32)
#pragma unroll
    for (int r = 0; r < 8; ++r) {
      int row = half * 8 + r;
      float hv0 = tanh_fast(acc0[r]);
      float hv1 = tanh_fast(acc1[r]);
      lds_h1[row * RNN_H + col0] = (__bf16)hv0;
      lds_h1[row * RNN_H + col1] = (__bf16)hv1;
      long obase = ((long)(brow0 + row) * RNN_T + t) * RNN_H;
      gout[obase + col0] = hv0;
      gout[obase + col1] = hv1;
    }
    __syncthreads();  // h1_new visible before next timestep
  }
}

extern "C" void kernel_launch(void* const* d_in, const int* in_sizes, int n_in,
                              void* d_out, int out_size, void* d_ws, size_t ws_size,
                              hipStream_t stream) {
  const float* x    = (const float*)d_in[0];
  const float* Wxh0 = (const float*)d_in[1];
  const float* Whh0 = (const float*)d_in[2];
  const float* bh0  = (const float*)d_in[3];
  const float* Wxh1 = (const float*)d_in[4];
  const float* Whh1 = (const float*)d_in[5];
  const float* bh1  = (const float*)d_in[6];
  float* out = (float*)d_out;

  // workspace: pre-transposed bf16 weights, [N][K] each (~1.75 MB total)
  unsigned short* WxT0 = (unsigned short*)d_ws;
  unsigned short* WhT0 = WxT0 + RNN_H * RNN_I;
  unsigned short* WxT1 = WhT0 + RNN_H * RNN_H;
  unsigned short* WhT1 = WxT1 + RNN_H * RNN_H;

  const int nI = RNN_H * RNN_I;
  const int nH = RNN_H * RNN_H;
  transpose_to_bf16<<<(nI + 255) / 256, 256, 0, stream>>>(Wxh0, WxT0, RNN_I, RNN_H);
  transpose_to_bf16<<<(nH + 255) / 256, 256, 0, stream>>>(Whh0, WhT0, RNN_H, RNN_H);
  transpose_to_bf16<<<(nH + 255) / 256, 256, 0, stream>>>(Wxh1, WxT1, RNN_H, RNN_H);
  transpose_to_bf16<<<(nH + 255) / 256, 256, 0, stream>>>(Whh1, WhT1, RNN_H, RNN_H);

  rnn_persistent<<<RNN_B / TILE_M, BLOCK, 0, stream>>>(
      x, bh0, bh1, WxT0, WhT0, WxT1, WhT1, out);
}
